// GNNConv_10763188043961
// MI455X (gfx1250) — matrix operand order repaired and checked by reading, not compiled, not run on hardware
//
#include <hip/hip_runtime.h>

typedef __attribute__((ext_vector_type(16))) _Float16 v16h;
typedef __attribute__((ext_vector_type(8)))  float    v8f;

#define N_NODES 100000
#define D 128
#define LN_EPS 1e-5f

// ---------------- kernel 1: in-degree histogram over targets (col) --------
__global__ __launch_bounds__(256) void deg_kernel(const int* __restrict__ col, int E,
                                                  int* __restrict__ deg) {
    int e = blockIdx.x * blockDim.x + threadIdx.x;
    if (e < E) atomicAdd(&deg[col[e]], 1);
}

// ---------------- kernel 2: convert W (f32, row-major [k][n]) -------------
// -> WhT (f16, transposed [n][k]) so each lane's B fragment (one column of W,
// 16 consecutive K) is a single contiguous 32B vector load.
__global__ __launch_bounds__(256) void wcvt_kernel(const float* __restrict__ W,
                                                   _Float16* __restrict__ WhT) {
    int i = blockIdx.x * blockDim.x + threadIdx.x;   // 0 .. 16383
    int k = i >> 7;          // row of W
    int n = i & 127;         // col of W
    WhT[(size_t)n * D + k] = (_Float16)W[i];
}

// ---------------- kernel 3: fused LayerNorm+ReLU -> WMMA GEMM -> g=dinv*h -
// One block = 16 nodes (rows). 8 waves; wave w computes the 16x16 output
// tile for columns [16w, 16w+16). K=128 consumed as 4 WMMA steps of K=32.
__global__ __launch_bounds__(256) void ln_gemm_kernel(
    const float* __restrict__ x, const float* __restrict__ gamma,
    const float* __restrict__ beta, const _Float16* __restrict__ WhT,
    const int* __restrict__ deg, float* __restrict__ g) {

    __shared__ _Float16 Atile[16][136];   // 16x128 f16 A tile, padded stride

    const int tid   = threadIdx.x;
    const int node0 = blockIdx.x * 16;

    // ---- Phase 1: LayerNorm + ReLU into LDS (16 threads per row, 8 cols each)
    {
        const int r   = tid >> 4;    // row 0..15
        const int c16 = tid & 15;    // col group 0..15
        const float* xr = x + (size_t)(node0 + r) * D + c16 * 8;
        float vals[8];
        float s = 0.f, ss = 0.f;
        #pragma unroll
        for (int i = 0; i < 8; ++i) { float v = xr[i]; vals[i] = v; s += v; ss += v * v; }
        // reduce across the 16 lanes owning this row (wave32: rows 2w / 2w+1)
        #pragma unroll
        for (int m = 8; m >= 1; m >>= 1) {
            s  += __shfl_xor(s,  m, 16);
            ss += __shfl_xor(ss, m, 16);
        }
        const float mean = s * (1.f / D);
        const float var  = ss * (1.f / D) - mean * mean;
        const float rstd = rsqrtf(var + LN_EPS);
        #pragma unroll
        for (int i = 0; i < 8; ++i) {
            const int cc = c16 * 8 + i;
            float v = (vals[i] - mean) * rstd * gamma[cc] + beta[cc];
            v = v > 0.f ? v : 0.f;
            Atile[r][cc] = (_Float16)v;
        }
    }
    __syncthreads();

    // ---- Phase 2: WMMA. wave w -> output columns n0..n0+15
    const int wave = tid >> 5;
    const int lane = tid & 31;
    const int n0   = wave * 16;
    const int m    = lane & 15;        // A row / B-C column selector
    const int hi   = (lane >= 16);
    const int n    = n0 + m;           // output column of this lane

    v8f c = {};
    #pragma unroll
    for (int kb = 0; kb < 4; ++kb) {
        // A fragment, this code's contiguous k order (same as B): lanes<16 hold K 0..15, lanes>=16 K 16..31
        const int kA = kb * 32 + (hi ? 16 : 0);
        v16h a;
        #pragma unroll
        for (int i = 0; i < 8; ++i) {
            a[i]     = Atile[m][kA + i];
            a[8 + i] = Atile[m][kA + 8 + i];
        }
        // B fragment, same contiguous k order as A: lane holds column n; lanes<16 K 0..15, lanes>=16 K 16..31.
        // WhT is [n][k], so this fragment is 16 contiguous halves (32B aligned).
        const int kB = kb * 32 + (hi ? 16 : 0);
        const v16h b = *(const v16h*)(WhT + (size_t)n * D + kB);

        c = __builtin_amdgcn_wmma_f32_16x16x32_f16(
                /*neg_a=*/false, a, /*neg_b=*/false, b,
                /*c_mod=*/(short)0, c, /*reuse_a=*/false, /*reuse_b=*/false);
    }

    // ---- write g = dinv[node] * h  (C/D layout: VGPR v -> row v (+8 for hi half))
    #pragma unroll
    for (int vv = 0; vv < 8; ++vv) {
        const int nd   = node0 + vv + (hi ? 8 : 0);
        const float dv = rsqrtf((float)(deg[nd] + 1));   // +1: self loop
        g[(size_t)nd * D + n] = c[vv] * dv;
    }
}

// ---------------- kernel 4: edge scatter: out[col] += g[row] --------------
// One wave per edge: float4 gather per lane + 4 fp32 L2 atomics per lane.
__global__ __launch_bounds__(256) void scatter_kernel(
    const int* __restrict__ row, const int* __restrict__ col, int E,
    const float* __restrict__ g, float* __restrict__ out) {

    const int wave = threadIdx.x >> 5;
    const int lane = threadIdx.x & 31;
    const int e    = blockIdx.x * 8 + wave;
    if (e >= E) return;                       // uniform per wave

    const int src = row[e];
    const int dst = col[e];
    const float4 v = ((const float4*)(g + (size_t)src * D))[lane];
    float* op = out + (size_t)dst * D + lane * 4;
    unsafeAtomicAdd(op + 0, v.x);
    unsafeAtomicAdd(op + 1, v.y);
    unsafeAtomicAdd(op + 2, v.z);
    unsafeAtomicAdd(op + 3, v.w);
}

// ---------------- kernel 5: out = dinv * (acc + g) + bias -----------------
__global__ __launch_bounds__(256) void finalize_kernel(
    const float* __restrict__ g, const int* __restrict__ deg,
    const float* __restrict__ bias, float* __restrict__ out) {

    const int idx  = blockIdx.x * blockDim.x + threadIdx.x;  // over N*32 float4s
    const int node = idx >> 5;
    const int q    = idx & 31;
    const float dv = rsqrtf((float)(deg[node] + 1));
    const float4 a  = ((const float4*)out)[idx];
    const float4 gg = ((const float4*)g)[idx];
    const float4 b  = ((const float4*)bias)[q];
    float4 r;
    r.x = dv * (a.x + gg.x) + b.x;
    r.y = dv * (a.y + gg.y) + b.y;
    r.z = dv * (a.z + gg.z) + b.z;
    r.w = dv * (a.w + gg.w) + b.w;
    ((float4*)out)[idx] = r;
}

extern "C" void kernel_launch(void* const* d_in, const int* in_sizes, int n_in,
                              void* d_out, int out_size, void* d_ws, size_t ws_size,
                              hipStream_t stream) {
    const float* x     = (const float*)d_in[0];
    const int*   ei    = (const int*)d_in[1];
    const float* gamma = (const float*)d_in[2];
    const float* beta  = (const float*)d_in[3];
    const float* W     = (const float*)d_in[4];
    const float* bias  = (const float*)d_in[5];
    float*       out   = (float*)d_out;

    const int E = in_sizes[1] / 2;
    const int* row = ei;           // sources j
    const int* col = ei + E;       // targets i

    // workspace layout: deg (N i32) | WhT (128x128 f16, transposed) | g (N x 128 f32)
    char* ws = (char*)d_ws;
    int* deg = (int*)ws;
    size_t off = ((size_t)N_NODES * 4 + 255) & ~(size_t)255;
    _Float16* WhT = (_Float16*)(ws + off);
    off += ((size_t)D * D * 2 + 255) & ~(size_t)255;
    float* g = (float*)(ws + off);

    hipMemsetAsync(deg, 0, (size_t)N_NODES * 4, stream);
    hipMemsetAsync(out, 0, (size_t)out_size * sizeof(float), stream);

    deg_kernel<<<(E + 255) / 256, 256, 0, stream>>>(col, E, deg);
    wcvt_kernel<<<(D * D) / 256, 256, 0, stream>>>(W, WhT);
    ln_gemm_kernel<<<N_NODES / 16, 256, 0, stream>>>(x, gamma, beta, WhT, deg, g);
    scatter_kernel<<<(E + 7) / 8, 256, 0, stream>>>(row, col, E, g, out);
    finalize_kernel<<<(N_NODES * 32 + 255) / 256, 256, 0, stream>>>(g, deg, bias, out);
}
